// TRT_53214644798206
// MI455X (gfx1250) — compile-verified, matrix operand order checked by reading, not run on hardware
//
#include <hip/hip_runtime.h>

// ---------------- model constants ----------------
#define NB   8
#define VV   25
#define TT   128
#define CC   256
#define NHH  8
#define DKK  32
#define DFFF 1024
#define ITER 4
#define NN   (NB * VV)   // 200
#define LY   (TT + 1)    // 129

__device__ __constant__ const float kEpsLN = 1e-6f;
__device__ __constant__ const float kEpsBN = 1e-5f;
__device__ __constant__ const float kSlope = 0.01f;
__device__ __constant__ const float kScale = 0.17677669529663687f; // 1/sqrt(32)

typedef __attribute__((ext_vector_type(16))) _Float16 v16h;
typedef __attribute__((ext_vector_type(8)))  _Float16 v8h;
typedef __attribute__((ext_vector_type(4)))  _Float16 v4h;
typedef __attribute__((ext_vector_type(8)))  float    v8f;

// =====================================================================
// WMMA GEMM:  Out[b] = W(MxK) * X[b](KxL) + bias   (+ relu | + residual)
// mode: 0 = bias only, 1 = bias+relu, 2 = bias+residual
// Block tile 128(M) x 64(L), K-step 32, 8 waves = 4(M) x 2(L).
// Each wave owns a 32x32 output tile: 2x2 fragment grid -> 4 independent
// v_wmma_f32_16x16x32_f16 per K-step from 8 contiguous ds_load_b128.
// A staged row-major, B staged column-major so every fragment is exactly
// two contiguous b128 LDS loads per lane (documented CDNA5 layouts).
// grid: (ceil(L/64), M/128, batch), block 256
// =====================================================================
template <int KDIM>
__global__ __launch_bounds__(256)
void gemm_wmma_kernel(const float* __restrict__ W, const float* __restrict__ X,
                      const float* __restrict__ bias, const float* __restrict__ Res,
                      float* __restrict__ Out, int M, int L, int mode)
{
    __shared__ __align__(16) _Float16 As[128][32];  // [m][k]  row-major  (8 KB)
    __shared__ __align__(16) _Float16 Bs[64][32];   // [l][k]  col-major  (4 KB)

    const int b  = blockIdx.z;
    const float* Xb = X + (size_t)b * KDIM * L;
    float*       Ob = Out + (size_t)b * M * L;
    const float* Rb = (mode == 2) ? (Res + (size_t)b * M * L) : nullptr;

    const int m0   = blockIdx.y * 128;
    const int l0   = blockIdx.x * 64;
    const int tid  = threadIdx.x;
    const int lane = tid & 31;
    const int wave = tid >> 5;
    const int wm   = wave & 3;   // 4 wave tiles along M (32 rows each)
    const int wn   = wave >> 2;  // 2 wave tiles along L (32 cols each)

    v8f acc00 = {}, acc01 = {}, acc10 = {}, acc11 = {};

    for (int k0 = 0; k0 < KDIM; k0 += 32) {
        // ---- stage A: 128x32 halves; 1024 float4 groups, 4 per thread ----
#pragma unroll
        for (int j = 0; j < 4; ++j) {
            int g  = tid + j * 256;
            int r  = g >> 3;
            int kq = (g & 7) << 2;
            float4 w4 = *(const float4*)&W[(size_t)(m0 + r) * KDIM + (k0 + kq)];
            v4h h;
            h[0] = (_Float16)w4.x; h[1] = (_Float16)w4.y;
            h[2] = (_Float16)w4.z; h[3] = (_Float16)w4.w;
            *(v4h*)&As[r][kq] = h;
        }
        // ---- stage B (transposed): 64 cols x 8 k-quads, 2 per thread ----
#pragma unroll
        for (int j = 0; j < 2; ++j) {
            int g   = tid + j * 256;
            int col = g & 63;            // consecutive tid -> consecutive l (coalesced)
            int kq  = (g >> 6) << 2;
            int l   = l0 + col;
            float b0 = 0.f, b1 = 0.f, b2 = 0.f, b3 = 0.f;
            if (l < L) {
                const float* xp = &Xb[(size_t)(k0 + kq) * L + l];
                b0 = xp[0];
                b1 = xp[(size_t)L];
                b2 = xp[(size_t)2 * L];
                b3 = xp[(size_t)3 * L];
            }
            v4h h;
            h[0] = (_Float16)b0; h[1] = (_Float16)b1;
            h[2] = (_Float16)b2; h[3] = (_Float16)b3;
            *(v4h*)&Bs[col][kq] = h;
        }
        if (k0 + 32 < KDIM)
            __builtin_prefetch(&W[(size_t)m0 * KDIM + (k0 + 32)], 0, 0);
        __syncthreads();

        // ---- fragments: two contiguous b128 LDS loads each ----
        // lanes 0-15: K halves {0..7,16..23}; lanes 16-31: {8..15,24..31}
        const int row  = lane & 15;
        const int koff = (lane >> 4) << 3;

        v8h a0lo = *(const v8h*)&As[wm * 32 + row][koff];
        v8h a0hi = *(const v8h*)&As[wm * 32 + row][koff + 16];
        v8h a1lo = *(const v8h*)&As[wm * 32 + 16 + row][koff];
        v8h a1hi = *(const v8h*)&As[wm * 32 + 16 + row][koff + 16];
        v8h b0lo = *(const v8h*)&Bs[wn * 32 + row][koff];
        v8h b0hi = *(const v8h*)&Bs[wn * 32 + row][koff + 16];
        v8h b1lo = *(const v8h*)&Bs[wn * 32 + 16 + row][koff];
        v8h b1hi = *(const v8h*)&Bs[wn * 32 + 16 + row][koff + 16];

        v16h af0, af1, bf0, bf1;
#pragma unroll
        for (int i = 0; i < 8; ++i) {
            af0[i] = a0lo[i]; af0[i + 8] = a0hi[i];
            af1[i] = a1lo[i]; af1[i + 8] = a1hi[i];
            bf0[i] = b0lo[i]; bf0[i + 8] = b0hi[i];
            bf1[i] = b1lo[i]; bf1[i + 8] = b1hi[i];
        }
        acc00 = __builtin_amdgcn_wmma_f32_16x16x32_f16(false, af0, false, bf0,
                                                       (short)0, acc00, false, false);
        acc01 = __builtin_amdgcn_wmma_f32_16x16x32_f16(false, af0, false, bf1,
                                                       (short)0, acc01, false, false);
        acc10 = __builtin_amdgcn_wmma_f32_16x16x32_f16(false, af1, false, bf0,
                                                       (short)0, acc10, false, false);
        acc11 = __builtin_amdgcn_wmma_f32_16x16x32_f16(false, af1, false, bf1,
                                                       (short)0, acc11, false, false);
        __syncthreads();
    }

    // C/D f32 layout: VGPR r -> (M = r + 8*(lane>=16), N = lane&15)
    const int mbase0 = m0 + wm * 32 + ((lane >> 4) << 3);
    const int mbase1 = mbase0 + 16;
    const int col0   = l0 + wn * 32 + (lane & 15);
    const int col1   = col0 + 16;

    auto emit = [&](const v8f& acc, int mb, int col) {
        if (col >= L) return;
#pragma unroll
        for (int r = 0; r < 8; ++r) {
            int m = mb + r;
            float val = acc[r] + bias[m];
            if (mode == 1)      val = fmaxf(val, 0.0f);
            else if (mode == 2) val += Rb[(size_t)m * L + col];
            Ob[(size_t)m * L + col] = val;
        }
    };
    emit(acc00, mbase0, col0);
    emit(acc01, mbase0, col1);
    emit(acc10, mbase1, col0);
    emit(acc11, mbase1, col1);
}

// =====================================================================
// LayerNorm over channel dim: x [N,C,T], one block per (n,t), block=C=256
// =====================================================================
__global__ __launch_bounds__(256)
void layernorm_kernel(const float* __restrict__ x, const float* __restrict__ g,
                      const float* __restrict__ b, float* __restrict__ y)
{
    const int nt = blockIdx.x;
    const int n = nt / TT, t = nt % TT;
    const int c = threadIdx.x;
    const size_t idx = ((size_t)n * CC + c) * TT + t;
    float v = x[idx];
    __shared__ float s1[256], s2[256];
    s1[c] = v; s2[c] = v * v;
    __syncthreads();
    for (int off = 128; off > 0; off >>= 1) {
        if (c < off) { s1[c] += s1[c + off]; s2[c] += s2[c + off]; }
        __syncthreads();
    }
    float m   = s1[0] * (1.0f / CC);
    float var = s2[0] * (1.0f / CC) - m * m;
    y[idx] = (v - m) * rsqrtf(var + kEpsLN) * g[c] + b[c];
}

// =====================================================================
// Train-mode BatchNorm over axes (0,2): x [Nb,C,L]; one block per channel.
// slope < 0 -> no activation; slope >= 0 -> fused leaky_relu(slope)
// =====================================================================
__global__ __launch_bounds__(256)
void batchnorm_kernel(const float* __restrict__ x, const float* __restrict__ g,
                      const float* __restrict__ b, float* __restrict__ y,
                      int Nb, int L, float slope)
{
    const int c = blockIdx.x, tid = threadIdx.x;
    const int total = Nb * L;
    float s = 0.0f, s2 = 0.0f;
    for (int i = tid; i < total; i += 256) {
        int n = i / L, l = i - n * L;
        float v = x[((size_t)n * CC + c) * L + l];
        s += v; s2 += v * v;
    }
    __shared__ float r1[256], r2[256];
    r1[tid] = s; r2[tid] = s2;
    __syncthreads();
    for (int off = 128; off > 0; off >>= 1) {
        if (tid < off) { r1[tid] += r1[tid + off]; r2[tid] += r2[tid + off]; }
        __syncthreads();
    }
    const float m   = r1[0] / (float)total;
    const float var = r2[0] / (float)total - m * m;
    const float inv = rsqrtf(var + kEpsBN);
    const float gg = g[c], bb = b[c];
    for (int i = tid; i < total; i += 256) {
        int n = i / L, l = i - n * L;
        size_t idx = ((size_t)n * CC + c) * L + l;
        float v = (x[idx] - m) * inv * gg + bb;
        if (slope >= 0.0f) v = (v > 0.0f) ? v : v * slope;
        y[idx] = v;
    }
}

// =====================================================================
// Windowed attention (window 3 + embed-token + relay-token = 5 keys).
// grid: N*NH, block: T=128 (thread = position t). out = xln + att.
// =====================================================================
__global__ __launch_bounds__(128)
void window_attn_kernel(const float* __restrict__ q, const float* __restrict__ k,
                        const float* __restrict__ v, const float* __restrict__ ke,
                        const float* __restrict__ ve, const float* __restrict__ krT,
                        const float* __restrict__ vrT, const float* __restrict__ xln,
                        float* __restrict__ out)
{
    const int nh = blockIdx.x;
    const int n = nh / NHH, h = nh % NHH;
    const int t = threadIdx.x;
    const size_t base = ((size_t)n * CC + h * DKK) * TT;

    float s0 = 0.f, s1 = 0.f, s2 = 0.f, s3 = 0.f, s4 = 0.f;
#pragma unroll 8
    for (int d = 0; d < DKK; ++d) {
        const size_t o = base + (size_t)d * TT + t;
        float qd = q[o];
        s1 += qd * k[o];
        if (t > 0)      s0 += qd * k[o - 1];
        if (t < TT - 1) s2 += qd * k[o + 1];
        s3 += qd * ke[o];
        s4 += qd * krT[(size_t)(h * DKK + d) * NN + n];
    }
    // zero-padded boundary keys give score 0 (still participate in softmax)
    s0 *= kScale; s1 *= kScale; s2 *= kScale; s3 *= kScale; s4 *= kScale;
    float mx = fmaxf(fmaxf(fmaxf(s0, s1), fmaxf(s2, s3)), s4);
    float e0 = __expf(s0 - mx), e1 = __expf(s1 - mx), e2 = __expf(s2 - mx);
    float e3 = __expf(s3 - mx), e4 = __expf(s4 - mx);
    float inv = 1.0f / (e0 + e1 + e2 + e3 + e4);
    e0 *= inv; e1 *= inv; e2 *= inv; e3 *= inv; e4 *= inv;

#pragma unroll 8
    for (int d = 0; d < DKK; ++d) {
        const size_t o = base + (size_t)d * TT + t;
        float vm = (t > 0)      ? v[o - 1] : 0.0f;
        float vp = (t < TT - 1) ? v[o + 1] : 0.0f;
        float att = e0 * vm + e1 * v[o] + e2 * vp + e3 * ve[o]
                  + e4 * vrT[(size_t)(h * DKK + d) * NN + n];
        out[o] = xln[o] + att;
    }
}

// =====================================================================
// Relay cross-attention over L=129 positions. grid N*NH, block 128.
// outT[c,n] = relayT[c,n] + att.
// =====================================================================
__global__ __launch_bounds__(128)
void relay_attn_kernel(const float* __restrict__ qT, const float* __restrict__ k2,
                       const float* __restrict__ v2, const float* __restrict__ relayT,
                       float* __restrict__ outT)
{
    const int nh = blockIdx.x;
    const int n = nh / NHH, h = nh % NHH;
    const int tid = threadIdx.x;
    __shared__ float sc[LY];
    __shared__ float qsh[DKK];
    __shared__ float sum_s;
    if (tid < DKK) qsh[tid] = qT[(size_t)(h * DKK + tid) * NN + n];
    __syncthreads();

    const size_t kbase = ((size_t)n * CC + h * DKK) * LY;
    for (int l = tid; l < LY; l += 128) {
        float s = 0.0f;
#pragma unroll 8
        for (int d = 0; d < DKK; ++d) s += qsh[d] * k2[kbase + (size_t)d * LY + l];
        sc[l] = s * kScale;
    }
    __syncthreads();
    if (tid == 0) {
        float mx = -1e30f;
        for (int l = 0; l < LY; ++l) mx = fmaxf(mx, sc[l]);
        float su = 0.0f;
        for (int l = 0; l < LY; ++l) { float e = __expf(sc[l] - mx); sc[l] = e; su += e; }
        sum_s = su;
    }
    __syncthreads();
    if (tid < DKK) {
        const float inv = 1.0f / sum_s;
        float acc = 0.0f;
        for (int l = 0; l < LY; ++l) acc += sc[l] * v2[kbase + (size_t)tid * LY + l];
        const size_t oi = (size_t)(h * DKK + tid) * NN + n;
        outT[oi] = relayT[oi] + acc * inv;
    }
}

// ---------------- layout / small kernels ----------------
__global__ __launch_bounds__(256)
void embs_kernel(const float* __restrict__ data, float* __restrict__ embs)
{
    int i = blockIdx.x * 256 + threadIdx.x;            // over N*C*T
    if (i >= NN * CC * TT) return;
    int t = i % TT, c = (i / TT) % CC, n = i / (TT * CC);
    int b = n / VV, v = n % VV;
    embs[i] = data[(((size_t)b * CC + c) * VV + v) * TT + t];
}

__global__ __launch_bounds__(256)
void copy_kernel(const float* __restrict__ src, float* __restrict__ dst, int n)
{
    int i = blockIdx.x * 256 + threadIdx.x;
    if (i < n) dst[i] = src[i];
}

__global__ __launch_bounds__(256)
void relay_init_kernel(const float* __restrict__ embs, float* __restrict__ relayT)
{
    int i = blockIdx.x * 256 + threadIdx.x;            // over C*N
    if (i >= CC * NN) return;
    int n = i % NN, c = i / NN;
    const size_t base = ((size_t)n * CC + c) * TT;
    float s = 0.0f;
    for (int t = 0; t < TT; ++t) s += embs[base + t];
    relayT[(size_t)c * NN + n] = s * (1.0f / TT);
}

__global__ __launch_bounds__(256)
void build_y_kernel(const float* __restrict__ relayT, const float* __restrict__ nodes,
                    float* __restrict__ y)
{
    int i = blockIdx.x * 256 + threadIdx.x;            // over N*C*LY
    if (i >= NN * CC * LY) return;
    int l = i % LY, c = (i / LY) % CC, n = i / (LY * CC);
    y[i] = (l == 0) ? relayT[(size_t)c * NN + n]
                    : nodes[((size_t)n * CC + c) * TT + (l - 1)];
}

// =====================================================================
// Host-side orchestration
// =====================================================================
extern "C" void kernel_launch(void* const* d_in, const int* in_sizes, int n_in,
                              void* d_out, int out_size, void* d_ws, size_t ws_size,
                              hipStream_t stream)
{
    (void)in_sizes; (void)n_in; (void)out_size; (void)ws_size;
    const float* data = (const float*)d_in[0];
    const float* ln_g = (const float*)d_in[1];
    const float* ln_b = (const float*)d_in[2];
    // tj: 3..16, tr: 17..30 (wq,bq,wk,bk,wv,bv,bn_g,bn_b,w1,b1,w2,b2,fbn_g,fbn_b)
    const float* tj[14]; const float* tr[14];
    for (int i = 0; i < 14; ++i) { tj[i] = (const float*)d_in[3 + i]; tr[i] = (const float*)d_in[17 + i]; }

    // ---- workspace carving (floats) ----
    const size_t SB  = (size_t)NN * CC * TT;   // 6.55M floats
    const size_t SBY = (size_t)NN * CC * LY;
    const size_t CN  = (size_t)CC * NN;
    const size_t HN  = (size_t)DFFF * NN;
    float* f      = (float*)d_ws;
    float* embs   = f;              // persistent
    float* nodes  = embs + SB;      // persistent
    float* xln    = nodes + SB;
    float* qb     = xln + SB;
    float* kb     = qb + SB;
    float* vb     = kb + SB;
    float* keb    = vb + SB;
    float* veb    = keb + SB;
    float* ret    = veb + SB;
    float* hbuf   = qb;             // reuse q..ke after attention (4*SB)
    float* obuf   = veb;            // reuse
    float* ybuf   = ret + SB;
    float* k2     = ybuf + SBY;
    float* v2     = k2 + SBY;
    float* relayT = v2 + SBY;
    float* qT     = relayT + CN;
    float* krT    = qT + CN;
    float* vrT    = krT + CN;
    float* tmpT   = vrT + CN;
    float* h2     = tmpT + CN;      // HN
    float* o2     = h2 + HN;        // CN

    auto gemm = [&](const float* W, const float* X, const float* bias, const float* Res,
                    float* Out, int M, int K, int L, int batch, int mode) {
        dim3 grid((L + 63) / 64, M / 128, batch);
        if (K == CC)
            gemm_wmma_kernel<CC><<<grid, 256, 0, stream>>>(W, X, bias, Res, Out, M, L, mode);
        else
            gemm_wmma_kernel<DFFF><<<grid, 256, 0, stream>>>(W, X, bias, Res, Out, M, L, mode);
    };

    const int totalE = NN * CC * TT;
    embs_kernel<<<(totalE + 255) / 256, 256, 0, stream>>>(data, embs);
    copy_kernel<<<(totalE + 255) / 256, 256, 0, stream>>>(embs, nodes, totalE);
    relay_init_kernel<<<(CC * NN + 255) / 256, 256, 0, stream>>>(embs, relayT);

    for (int it = 0; it < ITER; ++it) {
        // per-layer weight slices
        const float* wq = tj[0]  + (size_t)it * CC * CC;  const float* bq = tj[1]  + (size_t)it * CC;
        const float* wk = tj[2]  + (size_t)it * CC * CC;  const float* bk = tj[3]  + (size_t)it * CC;
        const float* wv = tj[4]  + (size_t)it * CC * CC;  const float* bv = tj[5]  + (size_t)it * CC;
        const float* bng = tj[6] + (size_t)it * CC;       const float* bnb = tj[7] + (size_t)it * CC;
        const float* w1 = tj[8]  + (size_t)it * DFFF * CC; const float* b1 = tj[9] + (size_t)it * DFFF;
        const float* w2 = tj[10] + (size_t)it * CC * DFFF; const float* b2 = tj[11] + (size_t)it * CC;
        const float* fg = tj[12] + (size_t)it * CC;       const float* fb = tj[13] + (size_t)it * CC;

        const float* rwq = tr[0]  + (size_t)it * CC * CC;  const float* rbq = tr[1]  + (size_t)it * CC;
        const float* rwk = tr[2]  + (size_t)it * CC * CC;  const float* rbk = tr[3]  + (size_t)it * CC;
        const float* rwv = tr[4]  + (size_t)it * CC * CC;  const float* rbv = tr[5]  + (size_t)it * CC;
        const float* rbng = tr[6] + (size_t)it * CC;       const float* rbnb = tr[7] + (size_t)it * CC;
        const float* rw1 = tr[8]  + (size_t)it * DFFF * CC; const float* rb1 = tr[9] + (size_t)it * DFFF;
        const float* rw2 = tr[10] + (size_t)it * CC * DFFF; const float* rb2 = tr[11] + (size_t)it * CC;
        const float* rfg = tr[12] + (size_t)it * CC;       const float* rfb = tr[13] + (size_t)it * CC;

        // ---- node path (tju) ----
        layernorm_kernel<<<NN * TT, 256, 0, stream>>>(nodes, ln_g + (size_t)it * CC,
                                                      ln_b + (size_t)it * CC, xln);
        gemm(wq, xln,  bq, nullptr, qb,  CC, CC, TT, NN, 0);
        gemm(wk, xln,  bk, nullptr, kb,  CC, CC, TT, NN, 0);
        gemm(wv, xln,  bv, nullptr, vb,  CC, CC, TT, NN, 0);
        gemm(wk, embs, bk, nullptr, keb, CC, CC, TT, NN, 0);   // K/V of embed tokens
        gemm(wv, embs, bv, nullptr, veb, CC, CC, TT, NN, 0);
        gemm(wk, relayT, bk, nullptr, krT, CC, CC, NN, 1, 0);  // K/V of relay token
        gemm(wv, relayT, bv, nullptr, vrT, CC, CC, NN, 1, 0);
        window_attn_kernel<<<NN * NHH, 128, 0, stream>>>(qb, kb, vb, keb, veb, krT, vrT, xln, ret);
        batchnorm_kernel<<<CC, 256, 0, stream>>>(ret, bng, bnb, ret, NN, TT, -1.0f);
        gemm(w1, ret,  b1, nullptr, hbuf, DFFF, CC, TT, NN, 1);     // relu
        gemm(w2, hbuf, b2, ret,     obuf, CC, DFFF, TT, NN, 2);     // + residual
        batchnorm_kernel<<<CC, 256, 0, stream>>>(obuf, fg, fb, nodes, NN, TT, kSlope); // + leaky

        // ---- relay path (tru) ----
        build_y_kernel<<<(NN * CC * LY + 255) / 256, 256, 0, stream>>>(relayT, nodes, ybuf);
        gemm(rwq, relayT, rbq, nullptr, qT, CC, CC, NN, 1, 0);
        gemm(rwk, ybuf,   rbk, nullptr, k2, CC, CC, LY, NN, 0);
        gemm(rwv, ybuf,   rbv, nullptr, v2, CC, CC, LY, NN, 0);
        relay_attn_kernel<<<NN * NHH, 128, 0, stream>>>(qT, k2, v2, relayT, tmpT);
        batchnorm_kernel<<<CC, 256, 0, stream>>>(tmpT, rbng, rbnb, tmpT, 1, NN, -1.0f);
        gemm(rw1, tmpT, rb1, nullptr, h2, DFFF, CC, NN, 1, 1);      // relu
        gemm(rw2, h2,   rb2, tmpT,    o2, CC, DFFF, NN, 1, 2);      // + residual
        batchnorm_kernel<<<CC, 256, 0, stream>>>(o2, rfg, rfb, relayT, 1, NN, kSlope); // + leaky
    }

    copy_kernel<<<(totalE + 255) / 256, 256, 0, stream>>>(nodes, (float*)d_out, totalE);
}
